// LSTM_autoencoder_27315992003082
// MI455X (gfx1250) — matrix operand order repaired and checked by reading, not compiled, not run on hardware
//
#include <hip/hip_runtime.h>

#define B_ 128
#define T_ 256
#define I_ 128
#define H_ 512
#define NWG_TOTAL 32
#define KTOT_MAX 1024

typedef _Float16 half_t;
typedef __attribute__((ext_vector_type(16))) _Float16 v16h;
typedef __attribute__((ext_vector_type(8)))  _Float16 v8h;
typedef __attribute__((ext_vector_type(8)))  float    v8f;

#define LOG2E 1.4426950408889634f

// sigmoid(x) = 1 / (1 + 2^(-x*log2e))   -> v_exp_f32 + v_rcp_f32 (hardware trans ops)
__device__ __forceinline__ float sigf(float x) {
  return __builtin_amdgcn_rcpf(1.0f + __builtin_amdgcn_exp2f(-LOG2E * x));
}
// tanh(x) = 2*sigmoid(2x) - 1
__device__ __forceinline__ float tanh_fast(float x) {
  return 2.0f * __builtin_amdgcn_rcpf(1.0f + __builtin_amdgcn_exp2f(-2.0f * LOG2E * x)) - 1.0f;
}

__device__ __forceinline__ v16h pack16(v8h lo, v8h hi) {
  v16h r;
#pragma unroll
  for (int i = 0; i < 8; ++i) { r[i] = lo[i]; r[i + 8] = hi[i]; }
  return r;
}

__device__ __forceinline__ v8f zero8() {
  v8f z;
#pragma unroll
  for (int i = 0; i < 8; ++i) z[i] = 0.0f;
  return z;
}

// A fragment (16 x 32, f16, row-major, row stride rs in halves), Arow0 = first row of strip.
// Contiguous k order (B matches): lanes 0-15 M=lane; 16 contiguous halves at K = kb + (lane>>4)*16.
__device__ __forceinline__ v16h load_a_g(const half_t* Arow0, long rs, int kb, int lane) {
  const half_t* p = Arow0 + (long)(lane & 15) * rs + kb + ((lane >> 4) << 4);
  return pack16(*(const v8h*)p, *(const v8h*)(p + 8));
}

// B fragment (32 x 16): weight row-major (N,K); lane n = n0+(lane&15),
// 16 contiguous halves at K = kb + (lane>>4)*16.
__device__ __forceinline__ v16h load_b_g(const half_t* Wrow0, long K, int kb, int lane) {
  const half_t* p = Wrow0 + (long)(lane & 15) * K + kb + ((lane >> 4) << 4);
  return pack16(*(const v8h*)p, *(const v8h*)(p + 8));
}

// ---------------------------------------------------------------------------
// Persistent fused 2-layer LSTM. 32 WGs: blocks 0..15 = layer0, 16..31 = layer1.
// Each WG owns 32 hidden columns (x4 gates = 128 z-columns), weights in LDS (f16).
// Software pipelined: at phase k, layer0 computes h0(k) while layer1 computes
// h1(k-1) from h0(k-1); one grid barrier per phase. c-state lives in registers.
// ---------------------------------------------------------------------------
extern "C" __global__ void __launch_bounds__(256)
lstm2_kernel(const half_t* __restrict__ in0, long in0_rs, long in0_ts, int K_in0,
             const float* __restrict__ Wih0, const float* __restrict__ Whh0, const float* __restrict__ b0,
             const float* __restrict__ Wih1, const float* __restrict__ Whh1, const float* __restrict__ b1,
             half_t* __restrict__ h0buf, half_t* __restrict__ h1buf,
             half_t* __restrict__ out_seq, unsigned* __restrict__ bar, int T)
{
  extern __shared__ half_t lds[];               // 128 x Ktot f16 weights
  float* bias = (float*)(lds + 128 * KTOT_MAX); // 128 f32 biases

  const int tid = threadIdx.x;
  const int wave = tid >> 5, lane = tid & 31;
  const int layer = blockIdx.x >> 4;
  const int wg = blockIdx.x & 15;
  const int col_base = wg * 32;

  const int Kin = layer ? H_ : K_in0;
  const int Ktot = Kin + H_;
  const float* Wih = layer ? Wih1 : Wih0;
  const float* Whh = layer ? Whh1 : Whh0;
  const float* bb  = layer ? b1 : b0;

  // Stage this WG's weight slice into LDS as f16: row nl = gate*32 + j, cols = [Wih | Whh]
  for (int idx = tid; idx < 128 * Ktot; idx += 256) {
    int nl = idx / Ktot, k = idx - nl * Ktot;
    int g = nl >> 5, j = nl & 31;
    int ng = g * H_ + col_base + j;
    float v = (k < Kin) ? Wih[(long)ng * Kin + k] : Whh[(long)ng * H_ + (k - Kin)];
    lds[(long)nl * Ktot + k] = (half_t)v;
  }
  for (int idx = tid; idx < 128; idx += 256) {
    int g = idx >> 5, j = idx & 31;
    bias[idx] = bb[g * H_ + col_base + j];
  }
  __syncthreads();

  const int mrow = wave * 16;         // each wave: 16 batch rows, all 128 z-columns
  float cst[2][8];
#pragma unroll
  for (int f = 0; f < 2; ++f)
#pragma unroll
    for (int r = 0; r < 8; ++r) cst[f][r] = 0.0f;

  for (int ph = 0; ph <= T; ++ph) {
    const bool active = layer ? (ph >= 1) : (ph < T);
    if (active) {
      const int t = layer ? (ph - 1) : ph;
      const half_t* xin; long xin_rs;
      const half_t* hprev; half_t* hwr;
      if (layer == 0) {
        xin = in0 + (long)t * in0_ts;              xin_rs = in0_rs;
        hprev = h0buf + ((ph + 1) & 1) * (128 * H_);   // h0(ph-1)
        hwr   = h0buf + (ph & 1) * (128 * H_);         // h0(ph)
      } else {
        xin = h0buf + ((ph + 1) & 1) * (128 * H_);     // h0(ph-1)
        xin_rs = H_;
        hprev = h1buf + (ph & 1) * (128 * H_);         // h1(ph-2)
        hwr   = h1buf + ((ph + 1) & 1) * (128 * H_);   // h1(ph-1)
      }

      v8f acc[8];
#pragma unroll
      for (int n = 0; n < 8; ++n) acc[n] = zero8();

      const half_t* abase_x = xin + (long)mrow * xin_rs;
      const half_t* abase_h = hprev + (long)mrow * H_;

      // K loop, x-input part (k in [0,Kin))
      for (int kb = 0; kb < Kin; kb += 32) {
        v16h a = load_a_g(abase_x, xin_rs, kb, lane);
#pragma unroll
        for (int n = 0; n < 8; ++n) {
          const half_t* p = &lds[(long)(n * 16 + (lane & 15)) * Ktot + kb + ((lane >> 4) << 4)];
          v16h bf = pack16(*(const v8h*)p, *(const v8h*)(p + 8));
          acc[n] = __builtin_amdgcn_wmma_f32_16x16x32_f16(false, a, false, bf,
                                                          (short)0, acc[n], false, false);
        }
      }
      // K loop, recurrent part (k in [Kin, Ktot))
      for (int kb = 0; kb < H_; kb += 32) {
        v16h a = load_a_g(abase_h, H_, kb, lane);
#pragma unroll
        for (int n = 0; n < 8; ++n) {
          const half_t* p = &lds[(long)(n * 16 + (lane & 15)) * Ktot + (Kin + kb) + ((lane >> 4) << 4)];
          v16h bf = pack16(*(const v8h*)p, *(const v8h*)(p + 8));
          acc[n] = __builtin_amdgcn_wmma_f32_16x16x32_f16(false, a, false, bf,
                                                          (short)0, acc[n], false, false);
        }
      }

      // LSTM cell: tiles n = gate*2 + f; C/D layout row = r + (lane>=16)*8, col = lane&15
      const int cl0 = lane & 15;
      const int rsel = (lane >> 4) * 8;
#pragma unroll
      for (int f = 0; f < 2; ++f) {
        const int cl = f * 16 + cl0;
        const float bi = bias[cl], bfv = bias[32 + cl], bg = bias[64 + cl], bo = bias[96 + cl];
#pragma unroll
        for (int r = 0; r < 8; ++r) {
          float zi = acc[0 + f][r] + bi;
          float zf = acc[2 + f][r] + bfv;
          float zg = acc[4 + f][r] + bg;
          float zo = acc[6 + f][r] + bo;
          float c = sigf(zf) * cst[f][r] + sigf(zi) * tanh_fast(zg);
          cst[f][r] = c;
          float h = sigf(zo) * tanh_fast(c);
          int row = mrow + r + rsel;
          int hc = col_base + cl;
          hwr[(long)row * H_ + hc] = (half_t)h;
          if (layer) out_seq[((long)row * T + t) * H_ + hc] = (half_t)h;
        }
      }
    }

    // grid-wide barrier (monotonic counter; zeroed by host memset each launch)
    __threadfence();
    __syncthreads();
    if (tid == 0) {
      __hip_atomic_fetch_add(bar, 1u, __ATOMIC_RELEASE, __HIP_MEMORY_SCOPE_AGENT);
      const unsigned tgt = (unsigned)(NWG_TOTAL * (ph + 1));
      while (__hip_atomic_load(bar, __ATOMIC_ACQUIRE, __HIP_MEMORY_SCOPE_AGENT) < tgt)
        __builtin_amdgcn_s_sleep(1);
    }
    __syncthreads();
  }
}

// ---------------------------------------------------------------------------
// Generic WMMA GEMM: C(M,N) = A(M,K)f16 @ W(N,K)^T f16 + bias.
// mode 0: relu -> f16 out ; mode 1: f32 out ; mode 2: f32 out shifted by one
// timestep (row b*T+t written to b*T+t+1, last t dropped).
// Grid: (N/128, M/128), 256 threads; wave w -> 16 rows x 128 cols (8 WMMA tiles).
// ---------------------------------------------------------------------------
extern "C" __global__ void __launch_bounds__(256)
gemm_kernel(const half_t* __restrict__ A, const half_t* __restrict__ W,
            const float* __restrict__ bias, int N, int K, int mode,
            half_t* __restrict__ o16, float* __restrict__ o32, int T)
{
  const int tid = threadIdx.x, wave = tid >> 5, lane = tid & 31;
  const long m0 = (long)blockIdx.y * 128 + wave * 16;
  const int n0 = blockIdx.x * 128;

  v8f acc[8];
#pragma unroll
  for (int n = 0; n < 8; ++n) acc[n] = zero8();

  const half_t* abase = A + m0 * K;
  for (int kb = 0; kb < K; kb += 32) {
    v16h a = load_a_g(abase, K, kb, lane);
#pragma unroll
    for (int n = 0; n < 8; ++n) {
      v16h b = load_b_g(W + (long)(n0 + n * 16) * K, K, kb, lane);
      acc[n] = __builtin_amdgcn_wmma_f32_16x16x32_f16(false, a, false, b,
                                                      (short)0, acc[n], false, false);
    }
  }

  const int rsel = (lane >> 4) * 8, cl = lane & 15;
#pragma unroll
  for (int n = 0; n < 8; ++n) {
    const int col = n0 + n * 16 + cl;
    const float bv = bias[col];
#pragma unroll
    for (int r = 0; r < 8; ++r) {
      long row = m0 + r + rsel;
      float v = acc[n][r] + bv;
      if (mode == 0) {
        v = fmaxf(v, 0.0f);
        o16[row * N + col] = (half_t)v;
      } else if (mode == 1) {
        o32[row * N + col] = v;
      } else {
        int t = (int)(row % T);
        if (t + 1 < T) o32[(row + 1) * (long)N + col] = v;
      }
    }
  }
}

extern "C" __global__ void cvt_f32_f16(const float* __restrict__ in, half_t* __restrict__ out, long n) {
  long i = (long)blockIdx.x * blockDim.x + threadIdx.x;
  if (i < n) out[i] = (half_t)in[i];
}

extern "C" __global__ void preds_t0_kernel(const float* __restrict__ x, float* __restrict__ preds) {
  int i = blockIdx.x * blockDim.x + threadIdx.x;
  if (i < B_ * I_) {
    int b = i / I_, ii = i - b * I_;
    preds[(long)b * T_ * I_ + ii] = x[((long)b * T_ + 1) * I_ + ii];
  }
}

extern "C" void kernel_launch(void* const* d_in, const int* in_sizes, int n_in,
                              void* d_out, int out_size, void* d_ws, size_t ws_size,
                              hipStream_t stream)
{
  (void)in_sizes; (void)n_in; (void)out_size; (void)ws_size;

  const float* x      = (const float*)d_in[0];
  const float* eWih0  = (const float*)d_in[1];
  const float* eWhh0  = (const float*)d_in[2];
  const float* eb0    = (const float*)d_in[3];
  const float* eWih1  = (const float*)d_in[4];
  const float* eWhh1  = (const float*)d_in[5];
  const float* eb1    = (const float*)d_in[6];
  const float* dWih0  = (const float*)d_in[7];
  const float* dWhh0  = (const float*)d_in[8];
  const float* db0    = (const float*)d_in[9];
  const float* dWih1  = (const float*)d_in[10];
  const float* dWhh1  = (const float*)d_in[11];
  const float* db1    = (const float*)d_in[12];
  const float* fc_W   = (const float*)d_in[13];
  const float* fc_b   = (const float*)d_in[14];
  const float* cls1_W = (const float*)d_in[15];
  const float* cls1_b = (const float*)d_in[16];
  const float* cls2_W = (const float*)d_in[17];
  const float* cls2_b = (const float*)d_in[18];

  float* dec_out   = (float*)d_out;                        // (B,T,I)
  float* preds_out = (float*)d_out + (long)B_ * T_ * I_;   // (B,T,I)

  // ---- workspace layout ----
  char* ws = (char*)d_ws;
  size_t off = 0;
  auto alloc = [&](size_t bytes) -> void* {
    void* p = ws + off; off = (off + bytes + 255) & ~(size_t)255; return p;
  };
  unsigned* bar   = (unsigned*)alloc(256);                             // two counters (bar[0], bar[16])
  half_t*   hste  = (half_t*)alloc(sizeof(half_t) * 4 * 2 * 128 * H_); // enc h0/h1, dec h0/h1 (dbl-buffered)
  size_t zero_bytes = off;                                             // zero bar + hidden state each call
  half_t* h0e = hste;
  half_t* h1e = hste + 2 * 128 * H_;
  half_t* h0d = hste + 4 * 128 * H_;
  half_t* h1d = hste + 6 * 128 * H_;

  const long MR = (long)B_ * T_;                      // 32768 rows
  half_t* x16   = (half_t*)alloc(sizeof(half_t) * MR * I_);
  half_t* enc1  = (half_t*)alloc(sizeof(half_t) * MR * H_);
  half_t* dec1  = (half_t*)alloc(sizeof(half_t) * MR * H_);
  half_t* Yrelu = (half_t*)alloc(sizeof(half_t) * MR * H_);
  half_t* c1_16 = (half_t*)alloc(sizeof(half_t) * H_ * H_);
  half_t* c2_16 = (half_t*)alloc(sizeof(half_t) * I_ * H_);
  half_t* fc16  = (half_t*)alloc(sizeof(half_t) * I_ * H_);

  hipMemsetAsync(d_ws, 0, zero_bytes, stream);

  // f32 -> f16 staging
  {
    long n = MR * I_;
    cvt_f32_f16<<<dim3((unsigned)((n + 255) / 256)), 256, 0, stream>>>(x, x16, n);
    n = (long)H_ * H_;
    cvt_f32_f16<<<dim3((unsigned)((n + 255) / 256)), 256, 0, stream>>>(cls1_W, c1_16, n);
    n = (long)I_ * H_;
    cvt_f32_f16<<<dim3((unsigned)((n + 255) / 256)), 256, 0, stream>>>(cls2_W, c2_16, n);
    cvt_f32_f16<<<dim3((unsigned)((n + 255) / 256)), 256, 0, stream>>>(fc_W, fc16, n);
  }

  const size_t SHMEM = 128 * KTOT_MAX * sizeof(half_t) + 128 * sizeof(float); // 256KB + 512B

  // Encoder: layer0 input = x16 (B,T,I); enc1 = layer-1 hidden sequence
  lstm2_kernel<<<NWG_TOTAL, 256, SHMEM, stream>>>(
      x16, (long)T_ * I_, (long)I_, I_,
      eWih0, eWhh0, eb0, eWih1, eWhh1, eb1,
      h0e, h1e, enc1, bar, T_);

  // preds = relu(enc @ cls1^T + b1) @ cls2^T + b2, shifted by one step
  gemm_kernel<<<dim3(H_ / 128, (unsigned)(MR / 128)), 256, 0, stream>>>(
      enc1, c1_16, cls1_b, H_, H_, 0, Yrelu, nullptr, T_);
  gemm_kernel<<<dim3(1, (unsigned)(MR / 128)), 256, 0, stream>>>(
      Yrelu, c2_16, cls2_b, I_, H_, 2, nullptr, preds_out, T_);
  preds_t0_kernel<<<dim3((B_ * I_ + 255) / 256), 256, 0, stream>>>(x, preds_out);

  // Decoder: constant input z = enc1[:, T-1, :] (timestep stride 0)
  lstm2_kernel<<<NWG_TOTAL, 256, SHMEM, stream>>>(
      enc1 + (long)(T_ - 1) * H_, (long)T_ * H_, 0L, H_,
      dWih0, dWhh0, db0, dWih1, dWhh1, db1,
      h0d, h1d, dec1, bar + 16, T_);

  // dec = dec_h1 @ fc_W^T + fc_b
  gemm_kernel<<<dim3(1, (unsigned)(MR / 128)), 256, 0, stream>>>(
      dec1, fc16, fc_b, I_, H_, 1, nullptr, dec_out, T_);
}
